// RoutingLayer_53412213293583
// MI455X (gfx1250) — matrix operand order repaired and checked by reading, not compiled, not run on hardware
//
#include <hip/hip_runtime.h>

typedef __attribute__((ext_vector_type(16))) _Float16 v16h;
typedef __attribute__((ext_vector_type(8)))  _Float16 v8h;
typedef __attribute__((ext_vector_type(4)))  _Float16 v4h;
typedef __attribute__((ext_vector_type(8)))  float    v8f;
typedef __attribute__((ext_vector_type(4)))  int      v4i;

#define DDIM   512
#define KCAPS  8
#define HDIM   64
#define NNB    32
#define ROWS   33           // 1 self + 32 neighbors
#define MT     3            // ceil(33/16) M-tiles
#define MROWS  48           // padded rows for WMMA
#define STR    520          // padded LDS row stride (halves) -> conflict-free frag reads
#define PNC    32           // cols per staged W1 panel
#define NGRP   16           // 512 / PNC panels

#define AS1 __attribute__((address_space(1)))
#define AS3 __attribute__((address_space(3)))

#if __has_builtin(__builtin_amdgcn_global_load_async_to_lds_b128)
#define USE_ASYNC_LDS 1
#else
#define USE_ASYNC_LDS 0
#endif

#if USE_ASYNC_LDS
#if __has_builtin(__builtin_amdgcn_s_wait_asynccnt)
#define ASYNC_WAIT(n) __builtin_amdgcn_s_wait_asynccnt(n)
#else
#define ASYNC_WAIT(n) asm volatile("s_wait_asynccnt %0" ::"i"(n) : "memory")
#endif
#else
#define ASYNC_WAIT(n)
#endif

// ---- W1 (f32) -> f16 workspace copy -------------------------------------
__global__ __launch_bounds__(256) void w1_to_f16_kernel(
    const float* __restrict__ w, _Float16* __restrict__ o, int n4) {
  int i = blockIdx.x * blockDim.x + threadIdx.x;
  if (i < n4) {
    float4 v = ((const float4*)w)[i];
    v4h h = { (_Float16)v.x, (_Float16)v.y, (_Float16)v.z, (_Float16)v.w };
    ((v4h*)o)[i] = h;
  }
}

// stage one 32-col W1 panel (32 rows x 512 halves) into LDS, 16B chunks
__device__ __forceinline__ void stage_panel(const _Float16* __restrict__ w16,
                                            _Float16* dst, int g, int t) {
  for (int task = t; task < PNC * 64; task += 256) {   // 8 chunks per thread/lane
    int row = task >> 6;
    int off = task & 63;
    const _Float16* src = w16 + ((long)(g * PNC + row) * DDIM) + off * 8;
#if USE_ASYNC_LDS
    __builtin_amdgcn_global_load_async_to_lds_b128(
        (AS1 v4i*)src, (AS3 v4i*)&dst[row * STR + off * 8], 0, 0);
#else
    *(float4*)&dst[row * STR + off * 8] = *(const float4*)src;
#endif
  }
}

// ---- fused projection + norm + routing, one workgroup per (b,n) ---------
__global__ __launch_bounds__(256) void routing_fused_kernel(
    const float*    __restrict__ selfv,     // [BN, 512]
    const float*    __restrict__ neighv,    // [BN, 32, 512]
    const float*    __restrict__ b1,        // [512]
    const _Float16* __restrict__ w16,       // [512, 512] row n = out-col, contiguous over k
    const int*      __restrict__ d_maxiter,
    float*          __restrict__ out)       // [BN, 512]
{
  __shared__ _Float16 As[MROWS * STR];          // 49,920 B  input rows, f16
  __shared__ _Float16 Bt[2][PNC * STR];         // 66,560 B  double-buffered W1 panel
  __shared__ _Float16 Zs[ROWS * DDIM];          // 33,792 B  normalized z (s + nz), f16
  __shared__ float    Us[DDIM];                 //  2,048 B  routing state u
  __shared__ float    Ps[NNB * KCAPS];          //  1,024 B  logits / softmax p

  const int  t    = threadIdx.x;
  const int  lane = t & 31;
  const int  wave = t >> 5;
  const long bn   = blockIdx.x;

  // ---- Stage A: self row 0, neighbor rows 1..32, converted f32->f16 ----
  for (int task = t; task < ROWS * (DDIM / 4); task += 256) {
    int row = task >> 7;          // /128
    int c4  = task & 127;
    const float4* src = (row == 0)
        ? ((const float4*)(selfv + bn * DDIM))
        : ((const float4*)(neighv + (bn * NNB + (long)(row - 1)) * DDIM));
    float4 v = src[c4];
    v4h h = { (_Float16)v.x, (_Float16)v.y, (_Float16)v.z, (_Float16)v.w };
    *(v4h*)&As[row * STR + c4 * 4] = h;
  }
  {   // zero pad rows 33..47 (WMMA reads them)
    unsigned* pz = (unsigned*)&As[ROWS * STR];
    for (int i = t; i < (MROWS - ROWS) * STR / 2; i += 256) pz[i] = 0u;
  }

  // prime panel 0 (async path: overlaps with the A-staging barrier below)
  stage_panel(w16, &Bt[0][0], 0, t);
  __syncthreads();

  // ---- GEMM: Z = relu(A @ W1^T + b1), 16 double-buffered panels of 32 cols
  for (int g = 0; g < NGRP; ++g) {
    if (g + 1 < NGRP) {
      stage_panel(w16, &Bt[(g + 1) & 1][0], g + 1, t);   // overlap with compute
      ASYNC_WAIT(8);    // panel g done (its 8 ops/lane precede panel g+1's in order)
    } else {
      ASYNC_WAIT(0);
    }
    __syncthreads();

    const _Float16* bcur = &Bt[g & 1][0];
    for (int tt = wave; tt < MT * (PNC / 16); tt += 8) {   // 6 tiles, waves 0..5
      const int mt    = tt % MT;
      const int ntl   = tt / MT;
      const int rowa  = mt * 16 + (lane & 15);
      const int nl    = ntl * 16 + (lane & 15);
      const int ahalf = (lane >= 16) ? 8 : 0;    // lanes 16-31: K = 8..15, 24..31
      const int bhalf = (lane >= 16) ? 8 : 0;    // lanes 16-31: K = 8..15, 24..31
      v8f c = {0.f, 0.f, 0.f, 0.f, 0.f, 0.f, 0.f, 0.f};
      #pragma unroll
      for (int k0 = 0; k0 < DDIM; k0 += 32) {
        v8h alo = *(const v8h*)&As[rowa * STR + k0 + ahalf];
        v8h ahi = *(const v8h*)&As[rowa * STR + k0 + 16 + ahalf];
        v8h blo = *(const v8h*)&bcur[nl * STR + k0 + bhalf];
        v8h bhi = *(const v8h*)&bcur[nl * STR + k0 + 16 + bhalf];
        v16h a = __builtin_shufflevector(alo, ahi, 0,1,2,3,4,5,6,7,8,9,10,11,12,13,14,15);
        v16h b = __builtin_shufflevector(blo, bhi, 0,1,2,3,4,5,6,7,8,9,10,11,12,13,14,15);
        c = __builtin_amdgcn_wmma_f32_16x16x32_f16(false, a, false, b, (short)0, c,
                                                   false, false);
      }
      const int   colz  = g * PNC + ntl * 16 + (lane & 15);
      const float bias  = b1[colz];
      const int   rbase = mt * 16 + ((lane >= 16) ? 8 : 0);
      #pragma unroll
      for (int r = 0; r < 8; ++r) {
        int row = rbase + r;
        if (row < ROWS)
          Zs[row * DDIM + colz] = (_Float16)fmaxf(c[r] + bias, 0.0f);
      }
    }
    __syncthreads();
  }

  // ---- l2norm each (row, k-group of 64) in place; rotation avoids bank conflicts
  for (int task = t; task < ROWS * KCAPS; task += 256) {
    int base = task * HDIM;
    float ssq = 0.f;
    for (int j = 0; j < HDIM; ++j) {
      int jj = (j + t) & (HDIM - 1);
      float v = (float)Zs[base + jj];
      ssq += v * v;
    }
    float inv = 1.0f / fmaxf(sqrtf(ssq), 1e-12f);
    for (int j = 0; j < HDIM; ++j) {
      int jj = (j + t) & (HDIM - 1);
      Zs[base + jj] = (_Float16)((float)Zs[base + jj] * inv);
    }
  }
  __syncthreads();

  // ---- dynamic routing (TAU = 1) ----
  const int max_iter = d_maxiter[0];
  for (int it = 0; it < max_iter; ++it) {
    if (it == 0) {
      Ps[t] = 1.0f / (float)KCAPS;              // softmax of zeros
    } else {
      // logits: p[m][k] = <nz[m][k], u[k]>
      const int m = t >> 3, k = t & 7;
      const _Float16* nzrow = &Zs[(m + 1) * DDIM + k * HDIM];
      const float*    urow  = &Us[k * HDIM];
      float acc = 0.f;
      for (int j = 0; j < HDIM; ++j) {
        int jj = (j + t) & (HDIM - 1);
        acc += (float)nzrow[jj] * urow[jj];
      }
      Ps[t] = acc;
      __syncthreads();
      if (t < NNB) {                            // softmax over K for neighbor t
        float l[KCAPS]; float mx = -1e30f;
        for (int k2 = 0; k2 < KCAPS; ++k2) { l[k2] = Ps[t * KCAPS + k2]; mx = fmaxf(mx, l[k2]); }
        float s = 0.f;
        for (int k2 = 0; k2 < KCAPS; ++k2) { l[k2] = expf(l[k2] - mx); s += l[k2]; }
        float invs = 1.0f / s;
        for (int k2 = 0; k2 < KCAPS; ++k2) Ps[t * KCAPS + k2] = l[k2] * invs;
      }
    }
    __syncthreads();
    // u[k][h] = s[k][h] + sum_m nz[m][k][h] * p[m][k]
    for (int task = t; task < DDIM; task += 256) {
      int k = task >> 6;
      float acc = (float)Zs[task];              // row 0 = s
      for (int m = 0; m < NNB; ++m)
        acc += (float)Zs[(m + 1) * DDIM + task] * Ps[m * KCAPS + k];
      Us[task] = acc;
    }
    __syncthreads();
    if (it < max_iter - 1) {
      if (t < KCAPS) {
        float ssq = 0.f;
        for (int j = 0; j < HDIM; ++j) { float v = Us[t * HDIM + j]; ssq += v * v; }
        float inv = 1.0f / fmaxf(sqrtf(ssq), 1e-12f);
        for (int j = 0; j < HDIM; ++j) Us[t * HDIM + j] *= inv;
      }
      __syncthreads();
    }
  }

  // ---- output: relu(u) ----
  for (int task = t; task < DDIM; task += 256)
    out[bn * DDIM + task] = fmaxf(Us[task], 0.0f);
}

extern "C" void kernel_launch(void* const* d_in, const int* in_sizes, int n_in,
                              void* d_out, int out_size, void* d_ws, size_t ws_size,
                              hipStream_t stream) {
  const float* selfv  = (const float*)d_in[0];   // [B,N,512]
  const float* neighv = (const float*)d_in[1];   // [B,N,32,512]
  const float* W1     = (const float*)d_in[2];   // [512,512]
  const float* b1     = (const float*)d_in[3];   // [512]
  const int*   maxit  = (const int*)d_in[4];
  float*       out    = (float*)d_out;

  _Float16* w16 = (_Float16*)d_ws;               // 512 KB

  const int wn4 = in_sizes[2] / 4;               // 65536
  w1_to_f16_kernel<<<(wn4 + 255) / 256, 256, 0, stream>>>(W1, w16, wn4);

  const int BN = in_sizes[0] / DDIM;             // 6400
  routing_fused_kernel<<<BN, 256, 0, stream>>>(selfv, neighv, b1, w16, maxit, out);
}
